// MultiHeadAttention_59356448031551
// MI455X (gfx1250) — matrix operand order repaired and checked by reading, not compiled, not run on hardware
//
#include <hip/hip_runtime.h>
#include <hip/hip_bf16.h>

#define B_  2
#define S_  2048
#define D_  1024
#define H_  16
#define DH_ 64

typedef __bf16 v16bf __attribute__((ext_vector_type(16)));
typedef float  v8f   __attribute__((ext_vector_type(8)));

union BFrag { v16bf v; unsigned short u[16]; uint4 q[2]; };

__device__ inline v8f wmma_bf16(v16bf a, v16bf b, v8f c) {
  // D = A(16x32 bf16) * B(32x16 bf16) + C(16x16 f32)
  return __builtin_amdgcn_wmma_f32_16x16x32_bf16(false, a, false, b, (short)0, c,
                                                 false, false);
}

__device__ inline unsigned short f2bf(float x) {  // RNE f32 -> bf16
  union { float f; unsigned u; } v; v.f = x;
  unsigned r = v.u + 0x7FFFu + ((v.u >> 16) & 1u);
  return (unsigned short)(r >> 16);
}

// A fragment 16x32 (MxK) bf16. lane&15 = row M; lane>>4 selects K sub-blocks:
//   u[0..7] <- K = half*8+0..7 ; u[8..15] <- K = 16+half*8+0..7
__device__ inline v16bf load_frag_a(const unsigned short* p, int pitch, int lane) {
  const int row = lane & 15, half = lane >> 4;
  const unsigned short* r = p + row * pitch;
  BFrag f;
  f.q[0] = *(const uint4*)(r + half * 8);
  f.q[1] = *(const uint4*)(r + 16 + half * 8);
  return f.v;
}

// B fragment 32x16 (KxN) bf16. lane&15 = col N; lane>>4 selects K sub-blocks:
//   u[0..7] <- K = half*8+0..7 ; u[8..15] <- K = 16+half*8+0..7
__device__ inline v16bf load_frag_b(const unsigned short* p, int col_pitch, int lane) {
  const int n = lane & 15, half = lane >> 4;
  const unsigned short* r = p + n * col_pitch;
  BFrag f;
  f.q[0] = *(const uint4*)(r + half * 8);
  f.q[1] = *(const uint4*)(r + 16 + half * 8);
  return f.v;
}

// Async 16B-per-lane copy global -> LDS (ASYNCcnt-tracked, no VGPR landing).
__device__ inline void async_cp_b128(unsigned lds_addr, const void* gaddr) {
  asm volatile("global_load_async_to_lds_b128 %0, %1, off"
               :: "v"(lds_addr), "v"((unsigned long long)(size_t)gaddr)
               : "memory");
}
__device__ inline void wait_async0() {
  asm volatile("s_wait_asynccnt 0x0" ::: "memory");
}

// ---------------------------------------------------------------------------
// Prep kernel 1: fp32 -> bf16 pack (vectorized), for activations.
// ---------------------------------------------------------------------------
__global__ __launch_bounds__(256)
void cvt_bf16_kernel(const float* __restrict__ in, unsigned short* __restrict__ out,
                     int n4) {
  int i = blockIdx.x * 256 + threadIdx.x;
  if (i >= n4) return;
  float4 v = ((const float4*)in)[i];
  union { unsigned short u[4]; uint2 d; } t;
  t.u[0] = f2bf(v.x); t.u[1] = f2bf(v.y); t.u[2] = f2bf(v.z); t.u[3] = f2bf(v.w);
  ((uint2*)out)[i] = t.d;
}

// ---------------------------------------------------------------------------
// Prep kernel 2: W[K,N] fp32 -> WT[N,K] bf16 (so GEMM B-frags read contiguous K).
// ---------------------------------------------------------------------------
__global__ __launch_bounds__(256)
void transpose_cvt_kernel(const float* __restrict__ W, unsigned short* __restrict__ WT) {
  __shared__ unsigned short tile[32][34];
  const int tx = threadIdx.x & 31, ty = threadIdx.x >> 5;  // 32 x 8
  const int k0 = blockIdx.x * 32, n0 = blockIdx.y * 32;
#pragma unroll
  for (int i = 0; i < 4; ++i) {
    int k = ty + i * 8;
    tile[tx][k] = f2bf(W[(size_t)(k0 + k) * D_ + n0 + tx]);  // coalesced along n
  }
  __syncthreads();
#pragma unroll
  for (int i = 0; i < 4; ++i) {
    int n = ty + i * 8;
    WT[(size_t)(n0 + n) * D_ + k0 + tx] = tile[n][tx];       // coalesced along k
  }
}

// ---------------------------------------------------------------------------
// QKV GEMM: C[4096,1024] = Xbf[4096,1024] * WT[1024,1024]^T, bf16 WMMA.
// Block tile 128x64, 8 waves (each 32x32), K-step 64, double-buffered LDS
// filled with global_load_async_to_lds_b128 (copies overlap WMMA).
// Output head-major bf16:
//   transpose_out==0 : out[((b*H+h)*S + s)*DH + dh]   (Q, K)
//   transpose_out==1 : out[((b*H+h)*DH + dh)*S + s]   (V^T for the PV matmul)
// ---------------------------------------------------------------------------
__device__ inline void gemm_stage(const unsigned short* __restrict__ Xbf,
                                  const unsigned short* __restrict__ WT,
                                  unsigned short (*Xl)[72], unsigned short (*Wl)[72],
                                  int m0, int n0, int kbase, int wave, int lane) {
  const int r4 = lane >> 3;   // 4 rows per async op (8 lanes x 16B = 128B row)
  const int ch = lane & 7;    // 16B chunk within the 128B row
#pragma unroll
  for (int j = 0; j < 4; ++j) {                       // X tile: 128 rows
    int row = wave * 16 + j * 4 + r4;
    async_cp_b128((unsigned)(size_t)&Xl[row][ch * 8],
                  Xbf + (size_t)(m0 + row) * D_ + kbase + ch * 8);
  }
#pragma unroll
  for (int j = 0; j < 2; ++j) {                       // WT tile: 64 rows
    int row = wave * 8 + j * 4 + r4;
    async_cp_b128((unsigned)(size_t)&Wl[row][ch * 8],
                  WT + (size_t)(n0 + row) * D_ + kbase + ch * 8);
  }
}

__global__ __launch_bounds__(256)
void qkv_gemm_kernel(const unsigned short* __restrict__ Xbf,
                     const unsigned short* __restrict__ WT,
                     unsigned short* __restrict__ out, int transpose_out) {
  __shared__ alignas(16) unsigned short Xlds[2][128][72];  // 128x64 bf16 + pad
  __shared__ alignas(16) unsigned short Wlds[2][64][72];   // 64x64 bf16 + pad

  const int tid = threadIdx.x, lane = tid & 31, wave = tid >> 5;
  const int wm = wave & 3, wn = wave >> 2;
  const int m0 = blockIdx.x * 128, n0 = blockIdx.y * 64;

  v8f acc[2][2] = {};
  const int KT = D_ / 64;

  gemm_stage(Xbf, WT, Xlds[0], Wlds[0], m0, n0, 0, wave, lane);
  for (int kt = 0; kt < KT; ++kt) {
    const int buf = kt & 1;
    wait_async0();               // my async copies for tile kt have landed
    __syncthreads();             // everyone's copies visible; prior reads done
    if (kt + 1 < KT)             // prefetch next tile while computing this one
      gemm_stage(Xbf, WT, Xlds[buf ^ 1], Wlds[buf ^ 1], m0, n0,
                 (kt + 1) * 64, wave, lane);
#pragma unroll
    for (int kc = 0; kc < 2; ++kc) {
      v16bf a0 = load_frag_a(&Xlds[buf][wm * 32][kc * 32],      72, lane);
      v16bf a1 = load_frag_a(&Xlds[buf][wm * 32 + 16][kc * 32], 72, lane);
      v16bf b0 = load_frag_b(&Wlds[buf][wn * 32][kc * 32],      72, lane);
      v16bf b1 = load_frag_b(&Wlds[buf][wn * 32 + 16][kc * 32], 72, lane);
      acc[0][0] = wmma_bf16(a0, b0, acc[0][0]);
      acc[0][1] = wmma_bf16(a0, b1, acc[0][1]);
      acc[1][0] = wmma_bf16(a1, b0, acc[1][0]);
      acc[1][1] = wmma_bf16(a1, b1, acc[1][1]);
    }
  }

  const int half = lane >> 4, nl = lane & 15;
#pragma unroll
  for (int sm = 0; sm < 2; ++sm)
#pragma unroll
    for (int sn = 0; sn < 2; ++sn)
#pragma unroll
      for (int r = 0; r < 8; ++r) {
        int gm = m0 + wm * 32 + sm * 16 + r + (half << 3);
        int gn = n0 + wn * 32 + sn * 16 + nl;
        int b = gm >> 11, s = gm & (S_ - 1);
        int h = gn >> 6,  dh = gn & (DH_ - 1);
        unsigned short v = f2bf(acc[sm][sn][r]);
        if (transpose_out)
          out[((size_t)(b * H_ + h) * DH_ + dh) * S_ + s] = v;
        else
          out[((size_t)(b * H_ + h) * S_ + s) * DH_ + dh] = v;
      }
}

// ---------------------------------------------------------------------------
// Causal flash attention, one wave per 16-row query tile.
// Q:[B,H,S,DH] bf16 ; K:[B,H,S,DH] bf16 ; VT:[B,H,DH,S] bf16 ; out f32 [B,S,D].
// scores = (Q·K^T)/8 with causal -inf (== ref's (qk + NEG_mask)/8 + re-mask).
// ---------------------------------------------------------------------------
__global__ __launch_bounds__(128)
void flash_attn_kernel(const unsigned short* __restrict__ Q,
                       const unsigned short* __restrict__ K,
                       const unsigned short* __restrict__ VT,
                       float* __restrict__ out) {
  __shared__ alignas(16) unsigned short p_lds[4][16][40];

  const int tid  = threadIdx.x;
  const int lane = tid & 31;
  const int wave = tid >> 5;
  const int bh = blockIdx.x;                 // b*H + h
  const int m0 = (blockIdx.y * 4 + wave) * 16;
  const int half = lane >> 4, nl = lane & 15;

  const unsigned short* q_base = Q  + (size_t)(bh * S_ + m0) * DH_;
  const unsigned short* k_base = K  + (size_t)bh * S_ * DH_;
  const unsigned short* v_base = VT + (size_t)bh * DH_ * S_;

  const v16bf A0 = load_frag_a(q_base,      DH_, lane);  // Q dh 0..31
  const v16bf A1 = load_frag_a(q_base + 32, DH_, lane);  // Q dh 32..63

  v8f o[4] = {};
  float row_max[8], row_sum[8];
#pragma unroll
  for (int r = 0; r < 8; ++r) { row_max[r] = -1e30f; row_sum[r] = 0.f; }

  const int nsteps = (m0 + 16 + 31) >> 5;   // 32-key steps up to causal bound
  for (int kt = 0; kt < nsteps; ++kt) {
    const int k0 = kt * 32;
    const unsigned short* kp = k_base + (size_t)k0 * DH_;

    v8f s0 = {}, s1 = {};
    s0 = wmma_bf16(A0, load_frag_b(kp,                 DH_, lane), s0);
    s0 = wmma_bf16(A1, load_frag_b(kp + 32,            DH_, lane), s0);
    s1 = wmma_bf16(A0, load_frag_b(kp + 16 * DH_,      DH_, lane), s1);
    s1 = wmma_bf16(A1, load_frag_b(kp + 16 * DH_ + 32, DH_, lane), s1);

    float p0[8], p1[8], alpha[8];
#pragma unroll
    for (int r = 0; r < 8; ++r) {
      const int m = m0 + r + (half << 3);
      float x0 = (k0 + nl      <= m) ? s0[r] * 0.125f : -1e30f;
      float x1 = (k0 + 16 + nl <= m) ? s1[r] * 0.125f : -1e30f;
      float t = fmaxf(x0, x1);
#pragma unroll
      for (int off = 1; off < 16; off <<= 1)
        t = fmaxf(t, __shfl_xor(t, off, 16));
      float nm = fmaxf(row_max[r], t);
      alpha[r] = __expf(row_max[r] - nm);
      row_max[r] = nm;
      p0[r] = __expf(x0 - nm);
      p1[r] = __expf(x1 - nm);
      float ps = p0[r] + p1[r];
#pragma unroll
      for (int off = 1; off < 16; off <<= 1)
        ps += __shfl_xor(ps, off, 16);
      row_sum[r] = row_sum[r] * alpha[r] + ps;
    }
#pragma unroll
    for (int t = 0; t < 4; ++t)
#pragma unroll
      for (int r = 0; r < 8; ++r)
        o[t][r] *= alpha[r];

    // Re-layout P (C layout) -> A fragment via LDS bounce (intra-wave only)
#pragma unroll
    for (int r = 0; r < 8; ++r) {
      p_lds[wave][r + (half << 3)][nl]      = f2bf(p0[r]);
      p_lds[wave][r + (half << 3)][16 + nl] = f2bf(p1[r]);
    }
    asm volatile("s_wait_dscnt 0x0" ::: "memory");  // DS RAW within the wave
    const v16bf PA = load_frag_a(&p_lds[wave][0][0], 40, lane);

    const unsigned short* vp = v_base + k0;
#pragma unroll
    for (int t = 0; t < 4; ++t)
      o[t] = wmma_bf16(PA, load_frag_b(vp + (size_t)(t * 16) * S_, S_, lane), o[t]);
  }

  const int b = bh >> 4, h = bh & 15;
#pragma unroll
  for (int r = 0; r < 8; ++r) {
    const float inv = 1.f / row_sum[r];
    const int m = m0 + r + (half << 3);
#pragma unroll
    for (int t = 0; t < 4; ++t)
      out[(size_t)(b * S_ + m) * D_ + h * DH_ + t * 16 + nl] = o[t][r] * inv;
  }
}

extern "C" void kernel_launch(void* const* d_in, const int* in_sizes, int n_in,
                              void* d_out, int out_size, void* d_ws, size_t ws_size,
                              hipStream_t stream) {
  const float* X[3] = { (const float*)d_in[0], (const float*)d_in[1],
                        (const float*)d_in[2] };                    // q, k, v
  const float* W[3] = { (const float*)d_in[4], (const float*)d_in[5],
                        (const float*)d_in[6] };                    // WQ, WK, WV
  // d_in[3] = attn_mask (causal, known statically) — unused.

  const size_t per = (size_t)B_ * H_ * S_ * DH_;   // bf16 elems per Q/K/V tensor
  unsigned short* ws  = (unsigned short*)d_ws;
  unsigned short* qkv[3] = { ws, ws + per, ws + 2 * per };   // 24 MB
  unsigned short* xbf = ws + 3 * per;                        // 8 MB  (reused x3)
  unsigned short* wt  = xbf + (size_t)B_ * S_ * D_;          // 2 MB  (reused x3)

  const int n4 = B_ * S_ * D_ / 4;
  dim3 gt(D_ / 32, D_ / 32);
  dim3 gg(B_ * S_ / 128, D_ / 64);
  for (int i = 0; i < 3; ++i) {   // stream order serializes reuse of xbf/wt
    cvt_bf16_kernel<<<n4 / 256, 256, 0, stream>>>(X[i], xbf, n4);
    transpose_cvt_kernel<<<gt, 256, 0, stream>>>(W[i], wt);
    qkv_gemm_kernel<<<gg, 256, 0, stream>>>(xbf, wt, qkv[i], i == 2 ? 1 : 0);
  }

  dim3 g2(B_ * H_, S_ / 64);
  flash_attn_kernel<<<g2, 128, 0, stream>>>(qkv[0], qkv[1], qkv[2],
                                            (float*)d_out);
}
